// Attention1_7421703488164
// MI455X (gfx1250) — matrix operand order repaired and checked by reading, not compiled, not run on hardware
//
#include <hip/hip_runtime.h>
#include <hip/hip_bf16.h>

typedef __attribute__((ext_vector_type(16))) _Float16 v16h;
typedef __attribute__((ext_vector_type(8)))  _Float16 v8h;
typedef __attribute__((ext_vector_type(8)))  float    v8f;

#define WMMA16(a, b, c) __builtin_amdgcn_wmma_f32_16x16x32_f16( \
    false, (a), false, (b), (short)0, (c), false, false)

// Build a 16-half A/B fragment from two 8-half LDS/global reads.
__device__ __forceinline__ v16h make_frag(const _Float16* p0, const _Float16* p1) {
    v8h lo = *reinterpret_cast<const v8h*>(p0);
    v8h hi = *reinterpret_cast<const v8h*>(p1);
    v16h f;
#pragma unroll
    for (int i = 0; i < 8; ++i) { f[i] = lo[i]; f[i + 8] = hi[i]; }
    return f;
}

__device__ __forceinline__ float redmax16(float v) {
    v = fmaxf(v, __shfl_xor(v, 1));
    v = fmaxf(v, __shfl_xor(v, 2));
    v = fmaxf(v, __shfl_xor(v, 4));
    v = fmaxf(v, __shfl_xor(v, 8));
    return v;
}
__device__ __forceinline__ float redsum16(float v) {
    v += __shfl_xor(v, 1);
    v += __shfl_xor(v, 2);
    v += __shfl_xor(v, 4);
    v += __shfl_xor(v, 8);
    return v;
}

// ---------------------------------------------------------------------------
// Kernel 1: qkv = x @ w_in^T + b_in, fused RoPE on q,k; q scaled by 1/8.
// Writes q/k/v as f16 in [b][head][n][64] layout.
// M=8192 (b*n), N=3072, K=1024. Tile 128x64x32, 8 waves (4x2), 2x2 per wave.
// ---------------------------------------------------------------------------
__global__ __launch_bounds__(256) void qkv_rope_kernel(
    const float* __restrict__ x, const float* __restrict__ w_in,
    const float* __restrict__ b_in, const float* __restrict__ freqs,
    _Float16* __restrict__ qs, _Float16* __restrict__ ks_,
    _Float16* __restrict__ vs) {
    __shared__ __align__(16) _Float16 Asl[128][40];
    __shared__ __align__(16) _Float16 Bsl[64][40];

    const int tid  = threadIdx.x;
    const int wave = tid >> 5, lane = tid & 31;
    const int ln = lane & 15, kh = lane >> 4;
    const int wm = wave >> 1, wn = wave & 1;
    const int m0 = blockIdx.y * 128, n0 = blockIdx.x * 64;

    const int arow = tid >> 1, acol = (tid & 1) * 16;
    const int brow = tid >> 2, bcol = (tid & 3) * 8;

    v8f acc[2][2] = {};

    for (int k0 = 0; k0 < 1024; k0 += 32) {
        __syncthreads();
        // stage A (x -> f16)
        const float4* ap = reinterpret_cast<const float4*>(
            x + (size_t)(m0 + arow) * 1024 + k0 + acol);
#pragma unroll
        for (int u = 0; u < 4; ++u) {
            float4 t = ap[u];
            _Float16* d = &Asl[arow][acol + u * 4];
            d[0] = (_Float16)t.x; d[1] = (_Float16)t.y;
            d[2] = (_Float16)t.z; d[3] = (_Float16)t.w;
        }
        // stage B (w_in -> f16)
        const float4* bp = reinterpret_cast<const float4*>(
            w_in + (size_t)(n0 + brow) * 1024 + k0 + bcol);
#pragma unroll
        for (int u = 0; u < 2; ++u) {
            float4 t = bp[u];
            _Float16* d = &Bsl[brow][bcol + u * 4];
            d[0] = (_Float16)t.x; d[1] = (_Float16)t.y;
            d[2] = (_Float16)t.z; d[3] = (_Float16)t.w;
        }
        __syncthreads();

        v16h af[2], bf[2];
#pragma unroll
        for (int i = 0; i < 2; ++i) {
            const int row = wm * 32 + i * 16 + ln;
            af[i] = make_frag(&Asl[row][kh * 8], &Asl[row][16 + kh * 8]);
        }
#pragma unroll
        for (int j = 0; j < 2; ++j) {
            const int col = wn * 32 + j * 16 + ln;
            bf[j] = make_frag(&Bsl[col][kh * 8], &Bsl[col][16 + kh * 8]);
        }
#pragma unroll
        for (int i = 0; i < 2; ++i)
#pragma unroll
            for (int j = 0; j < 2; ++j)
                acc[i][j] = WMMA16(af[i], bf[j], acc[i][j]);
    }

    // Epilogue: bias + RoPE + scatter to [b][h][n][64] f16.
    const int region = (n0 + wn * 32) >> 10;  // 0=q, 1=k, 2=v (uniform per block)
#pragma unroll
    for (int j = 0; j < 2; ++j) {
        const int col = n0 + wn * 32 + j * 16 + ln;
        const float bias = b_in[col];
        const int cr = col & 1023, head = cr >> 6, d = cr & 63;
#pragma unroll
        for (int i = 0; i < 2; ++i) {
#pragma unroll
            for (int r = 0; r < 8; ++r) {
                const int m = m0 + wm * 32 + i * 16 + r + kh * 8;
                const int pos = m & 2047, bsel = m >> 11;
                float val = acc[i][j][r] + bias;
                const size_t dst =
                    ((size_t)(bsel * 16 + head) * 2048 + pos) * 64 + d;
                if (region < 2) {
                    // freqs in [0, 2pi): fast hardware trig is plenty accurate
                    const float f = freqs[pos * 64 + d];
                    const float c = __cosf(f), s = __sinf(f);
                    const float partner = __shfl_xor(val, 1);
                    float rot = (col & 1) ? (val * c + partner * s)
                                          : (val * c - partner * s);
                    if (region == 0) qs[dst] = (_Float16)(rot * 0.125f);
                    else             ks_[dst] = (_Float16)rot;
                } else {
                    vs[dst] = (_Float16)val;
                }
            }
        }
    }
}

// ---------------------------------------------------------------------------
// Kernel 2: flash attention. One block per (b*h, 128-query block).
// 8 waves, each owns a 16-row strip; 64-key chunks; online softmax.
// ---------------------------------------------------------------------------
__global__ __launch_bounds__(256) void attn_kernel(
    const _Float16* __restrict__ qs, const _Float16* __restrict__ ks_,
    const _Float16* __restrict__ vs, const unsigned char* __restrict__ mask,
    _Float16* __restrict__ ao) {
    __shared__ __align__(16) _Float16 Ksl[64][72];
    __shared__ __align__(16) _Float16 VTsl[64][72];
    __shared__ __align__(16) _Float16 Psl[8][16][72];

    const int tid  = threadIdx.x;
    const int wave = tid >> 5, lane = tid & 31;
    const int ln = lane & 15, kh = lane >> 4;
    const int bh = blockIdx.y, b = bh >> 4, head = bh & 15;
    const int q0 = blockIdx.x * 128;

    const _Float16* qb = qs  + (size_t)bh * 2048 * 64;
    const _Float16* kb = ks_ + (size_t)bh * 2048 * 64;
    const _Float16* vb = vs  + (size_t)bh * 2048 * 64;

    // Q A-fragments straight from global (rows are contiguous 8-half runs).
    const int qrow = q0 + wave * 16 + ln;
    v16h qf[2];
#pragma unroll
    for (int kc = 0; kc < 2; ++kc)
        qf[kc] = make_frag(qb + (size_t)qrow * 64 + kc * 32 + kh * 8,
                           qb + (size_t)qrow * 64 + kc * 32 + 16 + kh * 8);

    v8f o[4] = {};
    float mrun[8], lrun[8];
#pragma unroll
    for (int r = 0; r < 8; ++r) { mrun[r] = -3.0e38f; lrun[r] = 0.0f; }

    for (int c0 = 0; c0 < 2048; c0 += 64) {
        __syncthreads();
        {   // stage K row-major, V transposed
            const int row = tid >> 2, seg = (tid & 3) * 16;
            const uint4* ksrc = reinterpret_cast<const uint4*>(
                kb + ((size_t)(c0 + row)) * 64 + seg);
            *reinterpret_cast<uint4*>(&Ksl[row][seg])     = ksrc[0];
            *reinterpret_cast<uint4*>(&Ksl[row][seg + 8]) = ksrc[1];
            const _Float16* vsrc = vb + ((size_t)(c0 + row)) * 64 + seg;
#pragma unroll
            for (int e = 0; e < 16; ++e) VTsl[seg + e][row] = vsrc[e];
        }
        __syncthreads();

        // S = Q @ K^T for this wave's 16x64 strip
        v8f s[4];
#pragma unroll
        for (int t = 0; t < 4; ++t) {
            const int key = t * 16 + ln;
            v8f z = {};
            v16h b0 = make_frag(&Ksl[key][kh * 8],      &Ksl[key][16 + kh * 8]);
            v16h b1 = make_frag(&Ksl[key][32 + kh * 8], &Ksl[key][32 + 16 + kh * 8]);
            z = WMMA16(qf[0], b0, z);
            s[t] = WMMA16(qf[1], b1, z);
            if (!mask[b * 2048 + c0 + key]) {
#pragma unroll
                for (int r = 0; r < 8; ++r) s[t][r] = -1.0e9f;
            }
        }

        // online softmax per row (row = r + 8*kh; 64 cols across lanes 0..15)
#pragma unroll
        for (int r = 0; r < 8; ++r) {
            float mx = fmaxf(fmaxf(s[0][r], s[1][r]), fmaxf(s[2][r], s[3][r]));
            mx = redmax16(mx);
            const float mnew  = fmaxf(mrun[r], mx);
            const float alpha = __expf(mrun[r] - mnew);
            float rs = 0.0f;
#pragma unroll
            for (int t = 0; t < 4; ++t) {
                float p = __expf(s[t][r] - mnew);
                s[t][r] = p;
                rs += p;
            }
            rs = redsum16(rs);
            lrun[r] = lrun[r] * alpha + rs;
            mrun[r] = mnew;
#pragma unroll
            for (int t = 0; t < 4; ++t) o[t][r] *= alpha;
        }

        // P: C-layout -> A-layout via wave-private LDS strip
#pragma unroll
        for (int t = 0; t < 4; ++t)
#pragma unroll
            for (int r = 0; r < 8; ++r)
                Psl[wave][r + kh * 8][t * 16 + ln] = (_Float16)s[t][r];
        __syncthreads();

        v16h pf[2];
#pragma unroll
        for (int kc = 0; kc < 2; ++kc)
            pf[kc] = make_frag(&Psl[wave][ln][kc * 32 + kh * 8],
                               &Psl[wave][ln][kc * 32 + 16 + kh * 8]);

        // O += P @ V
#pragma unroll
        for (int t = 0; t < 4; ++t) {
            const int dcol = t * 16 + ln;
            v16h b0 = make_frag(&VTsl[dcol][kh * 8],      &VTsl[dcol][16 + kh * 8]);
            v16h b1 = make_frag(&VTsl[dcol][32 + kh * 8], &VTsl[dcol][32 + 16 + kh * 8]);
            o[t] = WMMA16(pf[0], b0, o[t]);
            o[t] = WMMA16(pf[1], b1, o[t]);
        }
    }

    // normalize and write attn-out in [b][n][head*64+d] f16
#pragma unroll
    for (int r = 0; r < 8; ++r) {
        const float inv = 1.0f / lrun[r];
        const int qpos = q0 + wave * 16 + r + kh * 8;
#pragma unroll
        for (int t = 0; t < 4; ++t) {
            const int d = t * 16 + ln;
            ao[((size_t)b * 2048 + qpos) * 1024 + head * 64 + d] =
                (_Float16)(o[t][r] * inv);
        }
    }
}

// ---------------------------------------------------------------------------
// Kernel 3: out = attn @ w_out^T + b_out, zeroed where mask is false.
// M=8192, N=1024, K=1024. Same tiling as kernel 1.
// ---------------------------------------------------------------------------
__global__ __launch_bounds__(256) void proj_kernel(
    const _Float16* __restrict__ ao, const float* __restrict__ w_out,
    const float* __restrict__ b_out, const unsigned char* __restrict__ mask,
    float* __restrict__ out) {
    __shared__ __align__(16) _Float16 Asl[128][40];
    __shared__ __align__(16) _Float16 Bsl[64][40];

    const int tid  = threadIdx.x;
    const int wave = tid >> 5, lane = tid & 31;
    const int ln = lane & 15, kh = lane >> 4;
    const int wm = wave >> 1, wn = wave & 1;
    const int m0 = blockIdx.y * 128, n0 = blockIdx.x * 64;

    const int arow = tid >> 1, acol = (tid & 1) * 16;
    const int brow = tid >> 2, bcol = (tid & 3) * 8;

    v8f acc[2][2] = {};

    for (int k0 = 0; k0 < 1024; k0 += 32) {
        __syncthreads();
        const uint4* ap = reinterpret_cast<const uint4*>(
            ao + (size_t)(m0 + arow) * 1024 + k0 + acol);
        *reinterpret_cast<uint4*>(&Asl[arow][acol])     = ap[0];
        *reinterpret_cast<uint4*>(&Asl[arow][acol + 8]) = ap[1];
        const float4* bp = reinterpret_cast<const float4*>(
            w_out + (size_t)(n0 + brow) * 1024 + k0 + bcol);
#pragma unroll
        for (int u = 0; u < 2; ++u) {
            float4 t = bp[u];
            _Float16* d = &Bsl[brow][bcol + u * 4];
            d[0] = (_Float16)t.x; d[1] = (_Float16)t.y;
            d[2] = (_Float16)t.z; d[3] = (_Float16)t.w;
        }
        __syncthreads();

        v16h af[2], bf[2];
#pragma unroll
        for (int i = 0; i < 2; ++i) {
            const int row = wm * 32 + i * 16 + ln;
            af[i] = make_frag(&Asl[row][kh * 8], &Asl[row][16 + kh * 8]);
        }
#pragma unroll
        for (int j = 0; j < 2; ++j) {
            const int col = wn * 32 + j * 16 + ln;
            bf[j] = make_frag(&Bsl[col][kh * 8], &Bsl[col][16 + kh * 8]);
        }
#pragma unroll
        for (int i = 0; i < 2; ++i)
#pragma unroll
            for (int j = 0; j < 2; ++j)
                acc[i][j] = WMMA16(af[i], bf[j], acc[i][j]);
    }

#pragma unroll
    for (int j = 0; j < 2; ++j) {
        const int col = n0 + wn * 32 + j * 16 + ln;
        const float bias = b_out[col];
#pragma unroll
        for (int i = 0; i < 2; ++i) {
#pragma unroll
            for (int r = 0; r < 8; ++r) {
                const int m = m0 + wm * 32 + i * 16 + r + kh * 8;
                const int pos = m & 2047, bsel = m >> 11;
                const float val = acc[i][j][r] + bias;
                out[(size_t)m * 1024 + col] =
                    mask[bsel * 2048 + pos] ? val : 0.0f;
            }
        }
    }
}

extern "C" void kernel_launch(void* const* d_in, const int* in_sizes, int n_in,
                              void* d_out, int out_size, void* d_ws,
                              size_t ws_size, hipStream_t stream) {
    (void)in_sizes; (void)n_in; (void)out_size; (void)ws_size;
    const float*         x     = (const float*)d_in[0];
    const unsigned char* mask  = (const unsigned char*)d_in[1];
    const float*         freqs = (const float*)d_in[2];
    const float*         w_in  = (const float*)d_in[3];
    const float*         b_in  = (const float*)d_in[4];
    const float*         w_out = (const float*)d_in[5];
    const float*         b_out = (const float*)d_in[6];
    float*               out   = (float*)d_out;

    const size_t per = (size_t)64 * 2048 * 64;  // b*h * n * dh halves
    _Float16* qs = (_Float16*)d_ws;
    _Float16* ks = qs + per;
    _Float16* vs = ks + per;
    _Float16* ao = vs + per;

    qkv_rope_kernel<<<dim3(48, 64), 256, 0, stream>>>(
        x, w_in, b_in, freqs, qs, ks, vs);
    attn_kernel<<<dim3(16, 64), 256, 0, stream>>>(qs, ks, vs, mask, ao);
    proj_kernel<<<dim3(16, 64), 256, 0, stream>>>(ao, w_out, b_out, mask, out);
}
